// WavKANLinear_44564580663833
// MI455X (gfx1250) — matrix operand order repaired and checked by reading, not compiled, not run on hardware
//
#include <hip/hip_runtime.h>
#include <hip/hip_bf16.h>

// WavKAN linear: out[b,o] = silu(x) @ Wb^T  +  sum_{i,w} mexhat((x-t)/|s|) * ww
// B=256, IN=512, OUT=512, W=8

typedef __attribute__((ext_vector_type(16))) __bf16 v16bf;
typedef __attribute__((ext_vector_type(8)))  float  v8f;

#define B_SZ   256
#define IN_SZ  512
#define OUT_SZ 512
#define W_SZ   8

// ---------- helpers ----------
__device__ __forceinline__ unsigned short f32_to_bf16_rne(float f) {
    unsigned int u = __float_as_uint(f);
    unsigned int r = 0x7FFFu + ((u >> 16) & 1u);
    return (unsigned short)((u + r) >> 16);
}

__device__ __forceinline__ float silu_f(float x) {
    return x / (1.0f + __expf(-x));
}

// ---------- kernel 1: base GEMM via bf16 WMMA ----------
// C[b,o] = sum_i silu(x[b,i]) * Wb[o,i]
// Block tile 32x128, BK=64. 8 waves, each owns a 16x32 D tile:
//   per staging barrier: 2 K-steps x 2 N-subtiles = 4 WMMAs, A-frag reused.
#define BM 32
#define BN 128
#define BK 64
#define LDA 72   // padded LDS row stride (halfs): 144B rows -> 16B aligned, conflict-free
#define LDB 72

__global__ __launch_bounds__(256) void wavkan_base_gemm(
    const float* __restrict__ x,      // [B, IN]
    const float* __restrict__ wb,     // [OUT, IN]
    float* __restrict__ out)          // [B, OUT]
{
    __shared__ unsigned short As[BM * LDA];   // silu(x) tile, bf16, [b'][k']
    __shared__ unsigned short Bs[BN * LDB];   // Wb tile,      bf16, [o'][k']

    const int tid    = threadIdx.x;
    const int lane   = tid & 31;
    const int wid    = tid >> 5;       // 8 waves
    const int wm     = wid >> 2;       // 0..1  (16-row subtile)
    const int wn     = wid & 3;        // 0..3  (32-col subtile)
    const int bm0    = blockIdx.y * BM;
    const int bn0    = blockIdx.x * BN;
    const int half16 = lane >> 4;      // 0 or 1
    const int l16    = lane & 15;

    v8f acc0 = {};
    v8f acc1 = {};

    for (int k0 = 0; k0 < IN_SZ; k0 += BK) {
        // stage A: 32x64 floats, 8 per thread, coalesced rows
        #pragma unroll
        for (int j = 0; j < 8; ++j) {
            int idx = tid + j * 256;
            int r = idx >> 6, c = idx & 63;
            float v = x[(bm0 + r) * IN_SZ + k0 + c];
            As[r * LDA + c] = f32_to_bf16_rne(silu_f(v));
        }
        // stage B: 128x64 floats, 32 per thread, coalesced rows
        #pragma unroll
        for (int j = 0; j < 32; ++j) {
            int idx = tid + j * 256;
            int r = idx >> 6, c = idx & 63;
            Bs[r * LDB + c] = f32_to_bf16_rne(wb[(bn0 + r) * IN_SZ + k0 + c]);
        }
        __syncthreads();

        const unsigned short* rowA  = &As[(wm * 16 + l16) * LDA];
        const unsigned short* rowB0 = &Bs[(wn * 32 + l16) * LDB];
        const unsigned short* rowB1 = &Bs[(wn * 32 + 16 + l16) * LDB];

        #pragma unroll
        for (int ks = 0; ks < 2; ++ks) {
            const int kb = ks * 32;
            // A fragment (contiguous k order, same as B): k = kb + half*16 + e -> 32B contiguous
            union { v16bf v; uint4 q[2]; } fa, fb0, fb1;
            fa.q[0]  = *reinterpret_cast<const uint4*>(rowA + kb + half16 * 16);
            fa.q[1]  = *reinterpret_cast<const uint4*>(rowA + kb + half16 * 16 + 8);
            // B fragment (contiguous k order, same as A): k = kb + half*16 + e -> 32B contiguous
            fb0.q[0] = *reinterpret_cast<const uint4*>(rowB0 + kb + half16 * 16);
            fb0.q[1] = *reinterpret_cast<const uint4*>(rowB0 + kb + half16 * 16 + 8);
            fb1.q[0] = *reinterpret_cast<const uint4*>(rowB1 + kb + half16 * 16);
            fb1.q[1] = *reinterpret_cast<const uint4*>(rowB1 + kb + half16 * 16 + 8);

            acc0 = __builtin_amdgcn_wmma_f32_16x16x32_bf16(
                       false, fa.v, false, fb0.v, (short)0, acc0, false, false);
            acc1 = __builtin_amdgcn_wmma_f32_16x16x32_bf16(
                       false, fa.v, false, fb1.v, (short)0, acc1, false, false);
        }
        __syncthreads();
    }

    // D layout: VGPR r -> M = r + 8*(lane/16), N = lane%16
    const int m = bm0 + wm * 16 + half16 * 8;
    const int n0 = bn0 + wn * 32 + l16;
    #pragma unroll
    for (int r = 0; r < 8; ++r) {
        out[(m + r) * OUT_SZ + n0]      = acc0[r];
        out[(m + r) * OUT_SZ + n0 + 16] = acc1[r];
    }
}

// ---------- kernel 2: wavelet accumulation (VALU/exp bound) ----------
// block = one output column o, thread = batch row b.
// Per evaluation: d=x-t; d2=d*d; u=d2*inv2; E=exp2(d2*(-0.5*log2e*inv2));
//                 acc = fma(fma(-u,ww,ww), E, acc)   -> 6 VALU + 1 TRANS
#define CH 32   // i-chunk; CH*W_SZ == 256 -> one param element per thread per chunk

__device__ __forceinline__ float wav1(float xv, float t, float i2, float i2n,
                                      float ww, float acc) {
    float d  = xv - t;
    float d2 = d * d;
    float u  = d2 * i2;                          // z^2
    float E  = __builtin_amdgcn_exp2f(d2 * i2n); // exp(-0.5 z^2), native v_exp_f32
    return fmaf(fmaf(-u, ww, ww), E, acc);       // acc += ww*(1-u)*E
}

__global__ __launch_bounds__(256) void wavkan_wavelet(
    const float* __restrict__ x,      // [B, IN]
    const float* __restrict__ tr,     // [OUT, IN, W]
    const float* __restrict__ sc,     // [OUT, IN, W]
    const float* __restrict__ ww,     // [OUT, IN, W]
    float* __restrict__ out)          // [B, OUT] (accumulate)
{
    __shared__ alignas(16) float t_l [CH * W_SZ];
    __shared__ alignas(16) float i2_l[CH * W_SZ];
    __shared__ alignas(16) float in_l[CH * W_SZ];  // i2 * (-0.5*log2 e)
    __shared__ alignas(16) float w_l [CH * W_SZ];
    __shared__ float x_l [B_SZ][CH + 1];           // stride 33 -> conflict-free

    const int o   = blockIdx.x;
    const int tid = threadIdx.x;           // == b

    const float* trO = tr + (size_t)o * IN_SZ * W_SZ;
    const float* scO = sc + (size_t)o * IN_SZ * W_SZ;
    const float* wwO = ww + (size_t)o * IN_SZ * W_SZ;

    float acc = 0.0f;

    for (int i0 = 0; i0 < IN_SZ; i0 += CH) {
        // params for this i-chunk: contiguous 256 floats per tensor, 1/thread
        {
            int p = i0 * W_SZ + tid;
            float s   = fabsf(scO[p]) + 1e-8f;
            float inv = 1.0f / s;
            float i2  = inv * inv;
            t_l[tid]  = trO[p];
            i2_l[tid] = i2;
            in_l[tid] = i2 * -0.72134752044448170368f;  // -0.5 * log2(e) * inv2
            w_l[tid]  = wwO[p];
        }
        // x chunk: 256 rows x CH cols, coalesced
        #pragma unroll
        for (int j = 0; j < (B_SZ * CH) / 256; ++j) {
            int idx = tid + j * 256;
            int b = idx >> 5, c = idx & (CH - 1);
            x_l[b][c] = x[b * IN_SZ + i0 + c];
        }
        __syncthreads();

        #pragma unroll 2
        for (int ii = 0; ii < CH; ++ii) {
            float xv = x_l[tid][ii];
            const float4* T4 = reinterpret_cast<const float4*>(t_l)  + ii * 2;
            const float4* I4 = reinterpret_cast<const float4*>(i2_l) + ii * 2;
            const float4* N4 = reinterpret_cast<const float4*>(in_l) + ii * 2;
            const float4* W4 = reinterpret_cast<const float4*>(w_l)  + ii * 2;
            #pragma unroll
            for (int h = 0; h < 2; ++h) {
                float4 tt = T4[h];   // 16B LDS broadcast reads (ds_load_b128)
                float4 ii2 = I4[h];
                float4 nn = N4[h];
                float4 wq = W4[h];
                acc = wav1(xv, tt.x, ii2.x, nn.x, wq.x, acc);
                acc = wav1(xv, tt.y, ii2.y, nn.y, wq.y, acc);
                acc = wav1(xv, tt.z, ii2.z, nn.z, wq.z, acc);
                acc = wav1(xv, tt.w, ii2.w, nn.w, wq.w, acc);
            }
        }
        __syncthreads();
    }

    out[tid * OUT_SZ + o] += acc;   // GEMM kernel already wrote base_out
}

// ---------- launch ----------
extern "C" void kernel_launch(void* const* d_in, const int* in_sizes, int n_in,
                              void* d_out, int out_size, void* d_ws, size_t ws_size,
                              hipStream_t stream) {
    const float* x  = (const float*)d_in[0];   // [256,512]
    const float* tr = (const float*)d_in[1];   // [512,512,8]
    const float* sc = (const float*)d_in[2];   // [512,512,8]
    const float* ww = (const float*)d_in[3];   // [512,512,8]
    const float* wb = (const float*)d_in[4];   // [512,512]
    float* out = (float*)d_out;                // [256,512]

    dim3 g1(OUT_SZ / BN, B_SZ / BM);           // (4, 8)
    wavkan_base_gemm<<<g1, 256, 0, stream>>>(x, wb, out);
    wavkan_wavelet<<<OUT_SZ, 256, 0, stream>>>(x, tr, sc, ww, out);
}
